// DocREModel_18769007083641
// MI455X (gfx1250) — matrix operand order repaired and checked by reading, not compiled, not run on hardware
//
#include <hip/hip_runtime.h>
#include <stdint.h>

// ---------------------------------------------------------------------------
// Problem constants (from reference): n=4 docs, c=1024 seq, d=768, h=12 heads,
// E=32 entities, M=4 mentions, P=256 pairs, EMB=768, BLOCK=64, K=12, 97 labels
// ---------------------------------------------------------------------------
#define N_DOC   4
#define CLEN    1024
#define DMODEL  768
#define NHEAD   12
#define NENT    32
#define NMEN    4
#define NPAIR   256
#define RTOT    (N_DOC * NPAIR)     // 1024 rows
#define EMBD    768
#define BLK     64
#define KBIL    (EMBD * BLK)        // 49152
#define NOUT    97
#define NOUT_PAD 128                // padded so each wave covers 2 N-tiles

// ---------------------------------------------------------------------------
// WMMA fragment machinery (wave32, v_wmma_f32_16x16x32_bf16)
// ---------------------------------------------------------------------------
typedef __attribute__((ext_vector_type(16))) __bf16        v16bf;
typedef __attribute__((ext_vector_type(8)))  float         v8f;
typedef __attribute__((ext_vector_type(4)))  unsigned int  v4u;

union Frag16 {
  v16bf v;
  v4u   q[2];
  unsigned short h[16];
};

__device__ __forceinline__ unsigned short f2bf(float x) {
  unsigned int u = __float_as_uint(x);
  u += 0x7FFFu + ((u >> 16) & 1u);       // round-to-nearest-even
  return (unsigned short)(u >> 16);
}
__device__ __forceinline__ float bf2f(unsigned short s) {
  return __uint_as_float(((unsigned int)s) << 16);
}

__device__ __forceinline__ v8f wmma_bf16(v16bf a, v16bf b, v8f c) {
  return __builtin_amdgcn_wmma_f32_16x16x32_bf16(false, a, false, b, (short)0, c,
                                                 false, false);
}

// A fragment (16x32 bf16) from row-major A (K contiguous).
// ISA layout: lane<16 -> row M=lane, K {0..7, 16..23}; lane>=16 -> K {8..15, 24..31}
__device__ __forceinline__ v16bf load_a(const unsigned short* A, int lda, int m0, int k0) {
  int lane = threadIdx.x & 31;
  const unsigned short* p =
      A + (size_t)(m0 + (lane & 15)) * lda + k0 + ((lane & 16) ? 8 : 0);
  Frag16 f;
  f.q[0] = *(const v4u*)(p);        // K +0..7   (pairs per VGPR)
  f.q[1] = *(const v4u*)(p + 16);   // K +16..23
  return f.v;
}

// B fragment (32x16 bf16) from B stored [N][K] (K contiguous).
// ISA layout: N = lane%16; lane<16 -> K {0..7, 16..23}; lane>=16 -> K {8..15, 24..31}
__device__ __forceinline__ v16bf load_b(const unsigned short* B, int ldb, int n0, int k0) {
  int lane = threadIdx.x & 31;
  const unsigned short* p =
      B + (size_t)(n0 + (lane & 15)) * ldb + k0 + ((lane & 16) ? 8 : 0);
  Frag16 f;
  f.q[0] = *(const v4u*)(p);        // K +0..7
  f.q[1] = *(const v4u*)(p + 16);   // K +16..23
  return f.v;
}

// ---------------------------------------------------------------------------
// Stage 1: ent_emb[n,E,d] = logsumexp over M mentions of gathered embeddings
// ---------------------------------------------------------------------------
__global__ void k_ent_emb(const float* __restrict__ seq, const int* __restrict__ mpos,
                          float* __restrict__ ent_emb) {
  int b   = blockIdx.x;          // n*E = 128
  int doc = b >> 5, e = b & 31;
  int pos[NMEN];
#pragma unroll
  for (int m = 0; m < NMEN; ++m) pos[m] = mpos[(doc * NENT + e) * NMEN + m];
  for (int t = threadIdx.x; t < DMODEL; t += blockDim.x) {
    float x[NMEN];
#pragma unroll
    for (int m = 0; m < NMEN; ++m)
      x[m] = seq[((size_t)doc * CLEN + pos[m]) * DMODEL + t];
    float mx = x[0];
#pragma unroll
    for (int m = 1; m < NMEN; ++m) mx = fmaxf(mx, x[m]);
    float s = 0.f;
#pragma unroll
    for (int m = 0; m < NMEN; ++m) s += __expf(x[m] - mx);
    ent_emb[(size_t)b * DMODEL + t] = __logf(s) + mx;
  }
}

// ---------------------------------------------------------------------------
// Stage 2: ent_att[n,E,h,c] = mean over M of attention[n,h,pos,:]
// ---------------------------------------------------------------------------
__global__ void k_ent_att(const float* __restrict__ attn, const int* __restrict__ mpos,
                          float* __restrict__ ent_att) {
  int b   = blockIdx.x;                 // n*E*h = 1536
  int doc = b / (NENT * NHEAD);
  int rem = b % (NENT * NHEAD);
  int e   = rem / NHEAD;
  int hh  = rem % NHEAD;
  int pos[NMEN];
#pragma unroll
  for (int m = 0; m < NMEN; ++m) pos[m] = mpos[(doc * NENT + e) * NMEN + m];
  for (int cc = threadIdx.x; cc < CLEN; cc += blockDim.x) {
    float s = 0.f;
#pragma unroll
    for (int m = 0; m < NMEN; ++m)
      s += attn[(((size_t)doc * NHEAD + hh) * CLEN + pos[m]) * CLEN + cc];
    ent_att[(size_t)b * CLEN + cc] = s * 0.25f;
  }
}

// ---------------------------------------------------------------------------
// Stage 3: ht_att[n,P,c] = normalize_c( mean_h( h_att * t_att ) )  -> bf16
// ---------------------------------------------------------------------------
__global__ void k_htatt(const float* __restrict__ ent_att, const int* __restrict__ hts,
                        unsigned short* __restrict__ out_bf16) {
  __shared__ float partial[256];
  int b   = blockIdx.x;                 // n*P = 1024
  int doc = b >> 8;
  int eh  = hts[b * 2 + 0];
  int et  = hts[b * 2 + 1];
  const float* H = ent_att + ((size_t)(doc * NENT + eh)) * NHEAD * CLEN;
  const float* T = ent_att + ((size_t)(doc * NENT + et)) * NHEAD * CLEN;

  float vals[CLEN / 256];
  float lsum = 0.f;
#pragma unroll
  for (int it = 0; it < CLEN / 256; ++it) {
    int cc = threadIdx.x + it * 256;
    float s = 0.f;
#pragma unroll
    for (int hh = 0; hh < NHEAD; ++hh)
      s += H[hh * CLEN + cc] * T[hh * CLEN + cc];
    s *= (1.0f / NHEAD);
    vals[it] = s;
    lsum += s;
  }
  partial[threadIdx.x] = lsum;
  __syncthreads();
  for (int s = 128; s > 0; s >>= 1) {
    if ((int)threadIdx.x < s) partial[threadIdx.x] += partial[threadIdx.x + s];
    __syncthreads();
  }
  float inv = 1.0f / (partial[0] + 1e-30f);
#pragma unroll
  for (int it = 0; it < CLEN / 256; ++it) {
    int cc = threadIdx.x + it * 256;
    out_bf16[(size_t)b * CLEN + cc] = f2bf(vals[it] * inv);
  }
}

// ---------------------------------------------------------------------------
// Stage 4a: seqT[n][d][c] = bf16(seq[n][c][d])   (B operand of GEMM-1)
// ---------------------------------------------------------------------------
__global__ void k_seqT(const float* __restrict__ seq, unsigned short* __restrict__ seqT) {
  size_t idx = (size_t)blockIdx.x * blockDim.x + threadIdx.x;
  const size_t total = (size_t)N_DOC * DMODEL * CLEN;
  if (idx >= total) return;
  int doc = (int)(idx / ((size_t)DMODEL * CLEN));
  int rem = (int)(idx % ((size_t)DMODEL * CLEN));
  int dd  = rem / CLEN;
  int cc  = rem % CLEN;
  seqT[idx] = f2bf(seq[((size_t)doc * CLEN + cc) * DMODEL + dd]);
}

// ---------------------------------------------------------------------------
// Stage 4b: generic transpose+convert: out[Opad][K] = bf16(in[K][O]^T), zero pad
// ---------------------------------------------------------------------------
__global__ void k_transp(const float* __restrict__ in, unsigned short* __restrict__ out,
                         int K, int O, int Opad) {
  size_t idx = (size_t)blockIdx.x * blockDim.x + threadIdx.x;
  size_t total = (size_t)Opad * K;
  if (idx >= total) return;
  int o  = (int)(idx / K);
  int kk = (int)(idx % K);
  out[idx] = (o < O) ? f2bf(in[(size_t)kk * O + o]) : (unsigned short)0;
}

// ---------------------------------------------------------------------------
// Stage 5: A_head[r, 0:768] = bf16(ent_emb[doc, hts0]); A_tail likewise
// ---------------------------------------------------------------------------
__global__ void k_gather(const float* __restrict__ ent_emb, const int* __restrict__ hts,
                         unsigned short* __restrict__ A_head,
                         unsigned short* __restrict__ A_tail) {
  int r   = blockIdx.x;                 // 1024
  int doc = r >> 8;
  int eh  = hts[r * 2 + 0];
  int et  = hts[r * 2 + 1];
  for (int t = threadIdx.x; t < DMODEL; t += blockDim.x) {
    A_head[(size_t)r * (2 * DMODEL) + t] =
        f2bf(ent_emb[((size_t)doc * NENT + eh) * DMODEL + t]);
    A_tail[(size_t)r * (2 * DMODEL) + t] =
        f2bf(ent_emb[((size_t)doc * NENT + et) * DMODEL + t]);
  }
}

// ---------------------------------------------------------------------------
// GEMM-1 (WMMA): rs[doc] = ht_att[doc](256x1024) @ seq[doc](1024x768)
// result written as bf16 into A_head/A_tail columns [768:1536].
// 2x2 tiles per wave (32x32 C): fragments reused 2x, 4 independent WMMAs/step.
// grid = (M32=8, N32=24, doc=4), 32 threads (one wave)
// ---------------------------------------------------------------------------
__global__ void k_gemm_rs(const unsigned short* __restrict__ htatt,
                          const unsigned short* __restrict__ seqT,
                          unsigned short* __restrict__ A_head,
                          unsigned short* __restrict__ A_tail) {
  int m0  = blockIdx.x * 32;
  int n0  = blockIdx.y * 32;
  int doc = blockIdx.z;
  const unsigned short* A = htatt + (size_t)doc * NPAIR * CLEN;
  const unsigned short* B = seqT  + (size_t)doc * DMODEL * CLEN;
  v8f c00 = {}, c01 = {}, c10 = {}, c11 = {};
#pragma unroll 2
  for (int k0 = 0; k0 < CLEN; k0 += 32) {
    v16bf a0 = load_a(A, CLEN, m0, k0);
    v16bf a1 = load_a(A, CLEN, m0 + 16, k0);
    v16bf b0 = load_b(B, CLEN, n0, k0);
    v16bf b1 = load_b(B, CLEN, n0 + 16, k0);
    c00 = wmma_bf16(a0, b0, c00);
    c01 = wmma_bf16(a0, b1, c01);
    c10 = wmma_bf16(a1, b0, c10);
    c11 = wmma_bf16(a1, b1, c11);
  }
  int lane = threadIdx.x & 31;
#pragma unroll
  for (int tm = 0; tm < 2; ++tm) {
#pragma unroll
    for (int tn = 0; tn < 2; ++tn) {
      v8f c = tm ? (tn ? c11 : c10) : (tn ? c01 : c00);
      int col   = n0 + tn * 16 + (lane & 15);
      int rbase = m0 + tm * 16 + ((lane & 16) ? 8 : 0);
#pragma unroll
      for (int v = 0; v < 8; ++v) {
        size_t r = (size_t)doc * NPAIR + rbase + v;
        unsigned short val = f2bf(c[v]);
        A_head[r * (2 * DMODEL) + DMODEL + col] = val;
        A_tail[r * (2 * DMODEL) + DMODEL + col] = val;
      }
    }
  }
}

// ---------------------------------------------------------------------------
// GEMM-2 (WMMA): hs2/ts2 = tanh([hs|rs](1024x1536) @ W^T + b)  -> bf16
// 2x2 tiles per wave; grid = (M32=32, N32=24, z = head/tail)
// ---------------------------------------------------------------------------
__global__ void k_gemm_proj(const unsigned short* __restrict__ A_head,
                            const unsigned short* __restrict__ A_tail,
                            const unsigned short* __restrict__ WheadT,
                            const unsigned short* __restrict__ WtailT,
                            const float* __restrict__ b_head,
                            const float* __restrict__ b_tail,
                            unsigned short* __restrict__ hs2b,
                            unsigned short* __restrict__ ts2b) {
  int m0 = blockIdx.x * 32;
  int n0 = blockIdx.y * 32;
  int z  = blockIdx.z;
  const unsigned short* A    = z ? A_tail : A_head;
  const unsigned short* B    = z ? WtailT : WheadT;
  const float*          bias = z ? b_tail : b_head;
  unsigned short*       out  = z ? ts2b : hs2b;
  v8f c00 = {}, c01 = {}, c10 = {}, c11 = {};
#pragma unroll 2
  for (int k0 = 0; k0 < 2 * DMODEL; k0 += 32) {
    v16bf a0 = load_a(A, 2 * DMODEL, m0, k0);
    v16bf a1 = load_a(A, 2 * DMODEL, m0 + 16, k0);
    v16bf b0 = load_b(B, 2 * DMODEL, n0, k0);
    v16bf b1 = load_b(B, 2 * DMODEL, n0 + 16, k0);
    c00 = wmma_bf16(a0, b0, c00);
    c01 = wmma_bf16(a0, b1, c01);
    c10 = wmma_bf16(a1, b0, c10);
    c11 = wmma_bf16(a1, b1, c11);
  }
  int lane = threadIdx.x & 31;
#pragma unroll
  for (int tm = 0; tm < 2; ++tm) {
#pragma unroll
    for (int tn = 0; tn < 2; ++tn) {
      v8f c = tm ? (tn ? c11 : c10) : (tn ? c01 : c00);
      int col   = n0 + tn * 16 + (lane & 15);
      int rbase = m0 + tm * 16 + ((lane & 16) ? 8 : 0);
      float bv  = bias[col];
#pragma unroll
      for (int v = 0; v < 8; ++v)
        out[(size_t)(rbase + v) * DMODEL + col] = f2bf(tanhf(c[v] + bv));
    }
  }
}

// ---------------------------------------------------------------------------
// GEMM-3 (WMMA, fused outer product): logits = bl(1024x49152) @ W_bil + b_bil
// bl[r, k*4096+i*64+j] = hs2[r,k*64+i] * ts2[r,k*64+j] generated per fragment:
// within a 32-wide K chunk, (k,i) is constant -> 1 scalar + 32 contiguous ts2.
// The generated A fragment is shared across 2 N-tiles (halves VALU A-gen cost
// per WMMA, 2 independent WMMAs per step). grid = (M16=64, N32=4)
// ---------------------------------------------------------------------------
__global__ void k_gemm_bil(const unsigned short* __restrict__ hs2b,
                           const unsigned short* __restrict__ ts2b,
                           const unsigned short* __restrict__ WbilT,
                           const float* __restrict__ b_bil,
                           float* __restrict__ out) {
  int m0   = blockIdx.x * 16;
  int n0   = blockIdx.y * 32;
  int lane = threadIdx.x & 31;
  int r    = m0 + (lane & 15);
  const unsigned short* hrow = hs2b + (size_t)r * DMODEL;
  const unsigned short* trow = ts2b + (size_t)r * DMODEL + ((lane & 16) ? 8 : 0);
  v8f c0 = {}, c1 = {};
#pragma unroll 2
  for (int kk0 = 0; kk0 < KBIL; kk0 += 32) {
    int k   = kk0 >> 12;        // / 4096
    int rem = kk0 & 4095;
    int i   = rem >> 6;         // / 64
    int j0  = rem & 63;         // 0 or 32
    float b1 = bf2f(hrow[k * BLK + i]);
    const unsigned short* p = trow + k * BLK + j0;
    Frag16 t, a;
    t.q[0] = *(const v4u*)(p);
    t.q[1] = *(const v4u*)(p + 16);
#pragma unroll
    for (int e = 0; e < 16; ++e) a.h[e] = f2bf(b1 * bf2f(t.h[e]));
    v16bf bb0 = load_b(WbilT, KBIL, n0, kk0);
    v16bf bb1 = load_b(WbilT, KBIL, n0 + 16, kk0);
    c0 = wmma_bf16(a.v, bb0, c0);
    c1 = wmma_bf16(a.v, bb1, c1);
  }
  int rbase = m0 + ((lane & 16) ? 8 : 0);
#pragma unroll
  for (int tn = 0; tn < 2; ++tn) {
    v8f c   = tn ? c1 : c0;
    int col = n0 + tn * 16 + (lane & 15);
    if (col < NOUT) {
      float bv = b_bil[col];
#pragma unroll
      for (int v = 0; v < 8; ++v)
        out[(size_t)(rbase + v) * NOUT + col] = c[v] + bv;
    }
  }
}

// ---------------------------------------------------------------------------
// Host launcher
// ---------------------------------------------------------------------------
extern "C" void kernel_launch(void* const* d_in, const int* in_sizes, int n_in,
                              void* d_out, int out_size, void* d_ws, size_t ws_size,
                              hipStream_t stream) {
  (void)in_sizes; (void)n_in; (void)out_size; (void)ws_size;
  const float* seq   = (const float*)d_in[0];
  const float* attn  = (const float*)d_in[1];
  const int*   mpos  = (const int*)d_in[2];
  const int*   hts   = (const int*)d_in[3];
  const float* Whead = (const float*)d_in[4];
  const float* bhead = (const float*)d_in[5];
  const float* Wtail = (const float*)d_in[6];
  const float* btail = (const float*)d_in[7];
  const float* Wbil  = (const float*)d_in[8];
  const float* bbil  = (const float*)d_in[9];
  float* out = (float*)d_out;

  char* ws = (char*)d_ws;
  size_t off = 0;
  auto walloc = [&](size_t bytes) -> void* {
    void* p = ws + off;
    off += (bytes + 255) & ~(size_t)255;
    return p;
  };
  unsigned short* seqT    = (unsigned short*)walloc((size_t)N_DOC * DMODEL * CLEN * 2);
  unsigned short* WheadT  = (unsigned short*)walloc((size_t)DMODEL * 2 * DMODEL * 2);
  unsigned short* WtailT  = (unsigned short*)walloc((size_t)DMODEL * 2 * DMODEL * 2);
  unsigned short* WbilT   = (unsigned short*)walloc((size_t)NOUT_PAD * KBIL * 2);
  float*          ent_emb = (float*)walloc((size_t)N_DOC * NENT * DMODEL * 4);
  float*          ent_att = (float*)walloc((size_t)N_DOC * NENT * NHEAD * CLEN * 4);
  unsigned short* htatt   = (unsigned short*)walloc((size_t)RTOT * CLEN * 2);
  unsigned short* A_head  = (unsigned short*)walloc((size_t)RTOT * 2 * DMODEL * 2);
  unsigned short* A_tail  = (unsigned short*)walloc((size_t)RTOT * 2 * DMODEL * 2);
  unsigned short* hs2b    = (unsigned short*)walloc((size_t)RTOT * DMODEL * 2);
  unsigned short* ts2b    = (unsigned short*)walloc((size_t)RTOT * DMODEL * 2);

  // Stage 1-3: attention-side reductions
  k_ent_emb<<<N_DOC * NENT, 256, 0, stream>>>(seq, mpos, ent_emb);
  k_ent_att<<<N_DOC * NENT * NHEAD, 256, 0, stream>>>(attn, mpos, ent_att);
  k_htatt<<<RTOT, 256, 0, stream>>>(ent_att, hts, htatt);

  // Stage 4: bf16 operand staging (transposed B matrices)
  {
    size_t tot = (size_t)N_DOC * DMODEL * CLEN;
    k_seqT<<<(unsigned)((tot + 255) / 256), 256, 0, stream>>>(seq, seqT);
  }
  {
    size_t tot = (size_t)DMODEL * 2 * DMODEL;
    k_transp<<<(unsigned)((tot + 255) / 256), 256, 0, stream>>>(Whead, WheadT,
                                                                2 * DMODEL, DMODEL, DMODEL);
    k_transp<<<(unsigned)((tot + 255) / 256), 256, 0, stream>>>(Wtail, WtailT,
                                                                2 * DMODEL, DMODEL, DMODEL);
  }
  {
    size_t tot = (size_t)NOUT_PAD * KBIL;
    k_transp<<<(unsigned)((tot + 255) / 256), 256, 0, stream>>>(Wbil, WbilT,
                                                                KBIL, NOUT, NOUT_PAD);
  }

  // Stage 5: gather entity embeddings into concat buffers
  k_gather<<<RTOT, 256, 0, stream>>>(ent_emb, hts, A_head, A_tail);

  // Stage 6-8: WMMA GEMMs (one wave per block, wave32, multi-tile per wave)
  k_gemm_rs<<<dim3(NPAIR / 32, DMODEL / 32, N_DOC), 32, 0, stream>>>(htatt, seqT,
                                                                     A_head, A_tail);
  k_gemm_proj<<<dim3(RTOT / 32, DMODEL / 32, 2), 32, 0, stream>>>(
      A_head, A_tail, WheadT, WtailT, bhead, btail, hs2b, ts2b);
  k_gemm_bil<<<dim3(RTOT / 16, NOUT_PAD / 32), 32, 0, stream>>>(hs2b, ts2b, WbilT,
                                                                bbil, out);
}
